// Pipeline_80015240724847
// MI455X (gfx1250) — matrix operand order repaired and checked by reading, not compiled, not run on hardware
//
#include <hip/hip_runtime.h>

// ---------------------------------------------------------------------------
// CDNA5 (gfx1250) pipeline: MLP encoder -> concat -> 5x graph conv
//  * all activations f16 row-major; WMMA f32 accumulate
//  * weights pre-packed per launch into CDNA5 B-fragment layout (L2-resident)
//  * GEMM: A frags direct from global (2x b128/lane), B double-buffered into
//    LDS via global_load_async_to_lds_b128 (ASYNCcnt), 4 WMMAs/wave/K-step
//  * edge scatter: wave-per-edge f16 gather + f32 global atomics (HBM bound:
//    ~3.8 GB total @ 23.3 TB/s ~ 165 us floor; GEMMs ~20 GFLOP, negligible)
// ---------------------------------------------------------------------------

typedef __attribute__((ext_vector_type(16))) _Float16 v16h;
typedef __attribute__((ext_vector_type(8)))  _Float16 v8h;
typedef __attribute__((ext_vector_type(4)))  _Float16 v4h;
typedef __attribute__((ext_vector_type(8)))  float    v8f;

__device__ __forceinline__ void async_copy_b128(unsigned lds_off, const _Float16* g)
{
    // CDNA5 async global->LDS copy, 16B per enabled lane, tracked by ASYNCcnt
    asm volatile("global_load_async_to_lds_b128 %0, %1, off"
                 : : "v"(lds_off), "v"(g) : "memory");
}
__device__ __forceinline__ void wait_async0()
{
    asm volatile("s_wait_asynccnt 0x0" : : : "memory");
}

// ---------------------------------------------------------------------------
// Pack W (KxN f32 row-major) into B fragments [frag=nf*KT+kt][lane][16 halves]
// per ISA 16-bit B 32x16 layout: lane holds col nf*16+(lane&15);
// half s=2p+j = W[kt*32 + 8*(lane>>4) + s + (s>=8 ? 8 : 0)][col], zero-padded for col>=N.
// ---------------------------------------------------------------------------
__global__ __launch_bounds__(256)
void pack_w_kernel(const float* __restrict__ W, _Float16* __restrict__ Bp,
                   int K, int N, int nfrag)
{
    int frag = blockIdx.x * 8 + (threadIdx.x >> 5);
    if (frag >= nfrag) return;
    int lane  = threadIdx.x & 31;
    int KT    = K >> 5;
    int nf    = frag / KT;
    int kt    = frag - nf * KT;
    int col   = nf * 16 + (lane & 15);
    int kbase = kt * 32 + (lane >> 4) * 8;
    v16h out;
    #pragma unroll
    for (int p = 0; p < 8; ++p) {
        int k = kbase + 2 * p + (p >= 4 ? 8 : 0);
        float v0 = (col < N) ? W[(size_t)k * N + col]       : 0.0f;
        float v1 = (col < N) ? W[(size_t)(k + 1) * N + col] : 0.0f;
        out[2 * p]     = (_Float16)v0;
        out[2 * p + 1] = (_Float16)v1;
    }
    *(v16h*)(Bp + ((size_t)frag * 32 + lane) * 16) = out;
}

// ---------------------------------------------------------------------------
// C = act( A@W [+ A2@W2] + bias ).  A: MxK f16 row-major.  Bp: packed frags.
// Block = 256 thr = 8 waves; block tile 128M x 64N; wave = 16M x 4 frags.
// B double-buffered in LDS via async copies; A frags straight from global.
// ---------------------------------------------------------------------------
__global__ __launch_bounds__(256)
void wmma_gemm_async(const _Float16* __restrict__ A,  const _Float16* __restrict__ Bp,
                     const _Float16* __restrict__ A2, const _Float16* __restrict__ Bp2,
                     const float* __restrict__ bias,
                     float* __restrict__ Cf, _Float16* __restrict__ Ch,
                     int M, int N, int K, int relu)
{
    __shared__ __attribute__((aligned(16))) _Float16 Bsh[2][2048]; // 2 x 4KB

    const int tid   = threadIdx.x;
    const int lane  = tid & 31;
    const int wave  = tid >> 5;
    const int l15   = lane & 15;
    const int lhalf = lane >> 4;
    const int KT    = K >> 5;
    const int nf0   = blockIdx.x * 4;
    const int rowA  = blockIdx.y * 128 + wave * 16 + l15; // lane's A row (M)

    const int S = KT * (A2 ? 2 : 1);
    const unsigned lds0 = (unsigned)(size_t)(&Bsh[0][0]);

    const int sf = tid >> 6;   // staging: frag 0..3
    const int sc = tid & 63;   // staging: 16B chunk within frag

    auto stage = [&](int s) {  // issue async copy of B tiles for step s
        const _Float16* Bsrc = (s >= KT) ? Bp2 : Bp;
        int kt = (s >= KT) ? s - KT : s;
        const _Float16* g = Bsrc + ((size_t)(nf0 + sf) * KT + kt) * 512 + sc * 8;
        unsigned dst = lds0 + (unsigned)((s & 1) * 4096 + sf * 1024 + sc * 16);
        async_copy_b128(dst, g);
    };

    v8f acc0 = {}, acc1 = {}, acc2 = {}, acc3 = {};

    stage(0);
    for (int s = 0; s < S; ++s) {
        wait_async0();      // own wave's staged chunks landed
        __syncthreads();    // everyone's chunks landed
        if (s + 1 < S) stage(s + 1);   // overlap next B tile with compute

        const _Float16* Ap = (s >= KT) ? A2 : A;
        int kt = (s >= KT) ? s - KT : s;

        // A frag (ISA 16-bit A 16x32): two contiguous 16B chunks per lane
        const _Float16* ap = Ap + (size_t)rowA * K + kt * 32 + lhalf * 8;
        v8h alo = *(const v8h*)(ap);        // K = k0 + 8*lhalf + [0..7]
        v8h ahi = *(const v8h*)(ap + 16);   // K = k0 + 16 + 8*lhalf + [0..7]
        v16h a = __builtin_shufflevector(alo, ahi,
                    0,1,2,3,4,5,6,7,8,9,10,11,12,13,14,15);

        // B frags: contiguous 32B per lane from LDS (ds_load_b128 x2 each)
        const _Float16* bb = &Bsh[s & 1][lane * 16];
        v16h b0 = *(const v16h*)(bb);
        v16h b1 = *(const v16h*)(bb + 512);
        v16h b2 = *(const v16h*)(bb + 1024);
        v16h b3 = *(const v16h*)(bb + 1536);

        acc0 = __builtin_amdgcn_wmma_f32_16x16x32_f16(false, a, false, b0, (short)0, acc0, false, false);
        acc1 = __builtin_amdgcn_wmma_f32_16x16x32_f16(false, a, false, b1, (short)0, acc1, false, false);
        acc2 = __builtin_amdgcn_wmma_f32_16x16x32_f16(false, a, false, b2, (short)0, acc2, false, false);
        acc3 = __builtin_amdgcn_wmma_f32_16x16x32_f16(false, a, false, b3, (short)0, acc3, false, false);

        __syncthreads();    // all reads of Bsh[s&1] done before it is restaged
    }

    // epilogue: C/D layout — VGPR r holds M = r + 8*lhalf, N = lane&15
    const int rowBase = blockIdx.y * 128 + wave * 16 + 8 * lhalf;
    v8f accs[4] = {acc0, acc1, acc2, acc3};
    #pragma unroll
    for (int f = 0; f < 4; ++f) {
        int col = (nf0 + f) * 16 + l15;
        if (col < N) {
            float bv = bias ? bias[col] : 0.0f;
            #pragma unroll
            for (int r = 0; r < 8; ++r) {
                int row = rowBase + r;
                float v = accs[f][r] + bv;
                if (relu) v = fmaxf(v, 0.0f);
                if (Ch) Ch[(size_t)row * N + col] = (_Float16)v;
                else    Cf[(size_t)row * N + col] = v;
            }
        }
    }
}

// f32 -> f16 row-major (n multiple of 4)
__global__ __launch_bounds__(256)
void f32_to_f16_kernel(const float* __restrict__ in, _Float16* __restrict__ out, long n)
{
    long i = ((long)blockIdx.x * blockDim.x + threadIdx.x) * 4;
    if (i >= n) return;
    float4 v = *(const float4*)(in + i);
    v4h o = {(_Float16)v.x, (_Float16)v.y, (_Float16)v.z, (_Float16)v.w};
    *(v4h*)(out + i) = o;
}

// feat[node, 0:120]=enc(f16), [120:123]=pos, [123:128]=u[batch[node]]
__global__ __launch_bounds__(256)
void assemble_feat_kernel(const _Float16* __restrict__ enc, const float* __restrict__ pos,
                          const float* __restrict__ u, const int* __restrict__ batch,
                          _Float16* __restrict__ feat, int n)
{
    int t = blockIdx.x * blockDim.x + threadIdx.x;
    if (t >= n * 128) return;
    int node = t >> 7, f = t & 127;
    _Float16 v;
    if (f < 120)      v = enc[node * 120 + f];
    else if (f < 123) v = (_Float16)pos[node * 3 + (f - 120)];
    else              v = (_Float16)u[batch[node] * 5 + (f - 123)];
    feat[t] = v;
}

// One wave per edge; lane L gathers 4 halves (b64), scatter-adds 4 f32 atomics.
__global__ __launch_bounds__(256)
void edge_scatter_kernel(const _Float16* __restrict__ h, const float* __restrict__ ew,
                         const int* __restrict__ src, const int* __restrict__ dst,
                         float* __restrict__ agg, int E)
{
    int edge = blockIdx.x * 8 + (threadIdx.x >> 5);
    int lane = threadIdx.x & 31;
    if (edge >= E) return;
    int   s = src[edge];
    int   d = dst[edge];
    float w = ew[edge];
    v4h v = *(const v4h*)(h + (size_t)s * 128 + lane * 4);   // global_load_b64
    float* ap = agg + (size_t)d * 128 + lane * 4;
    atomicAdd(ap + 0, (float)v[0] * w);
    atomicAdd(ap + 1, (float)v[1] * w);
    atomicAdd(ap + 2, (float)v[2] * w);
    atomicAdd(ap + 3, (float)v[3] * w);
}

// ---------------------------------------------------------------------------
extern "C" void kernel_launch(void* const* d_in, const int* in_sizes, int n_in,
                              void* d_out, int out_size, void* d_ws, size_t ws_size,
                              hipStream_t stream)
{
    (void)in_sizes; (void)n_in; (void)out_size; (void)ws_size;

    const float* x        = (const float*)d_in[0];
    const float* u        = (const float*)d_in[1];
    const float* pos      = (const float*)d_in[2];
    const float* ew       = (const float*)d_in[3];
    const float* w1       = (const float*)d_in[4];
    const float* b1       = (const float*)d_in[5];
    const float* w2       = (const float*)d_in[6];
    const float* b2       = (const float*)d_in[7];
    const float* w3       = (const float*)d_in[8];
    const float* b3       = (const float*)d_in[9];
    const float* w4       = (const float*)d_in[10];
    const float* b4       = (const float*)d_in[11];
    const float* rel_w_h  = (const float*)d_in[12];  // (4,128,128)
    const float* rel_b_h  = (const float*)d_in[13];  // (4,128)
    const float* root_w_h = (const float*)d_in[14];  // (4,128,128)
    const float* rel_w_f  = (const float*)d_in[15];  // (128,120)
    const float* rel_b_f  = (const float*)d_in[16];  // (120)
    const float* root_w_f = (const float*)d_in[17];  // (128,120)
    const int*   eidx     = (const int*)d_in[18];    // (2,E)
    const int*   batch    = (const int*)d_in[19];    // (N)

    const int Nn = 16384;
    const int E_ = 1048576;

    // ---- workspace carve-out
    char* ws = (char*)d_ws;
    size_t o = 0;
    _Float16* xh   = (_Float16*)(ws + o); o += (size_t)Nn * 512 * 2;  // 16 MB
    _Float16* act0 = (_Float16*)(ws + o); o += (size_t)Nn * 512 * 2;  // 16 MB
    _Float16* act1 = (_Float16*)(ws + o); o += (size_t)Nn * 512 * 2;  // 16 MB
    _Float16* feat = (_Float16*)(ws + o); o += (size_t)Nn * 128 * 2;  //  4 MB
    _Float16* hnxt = (_Float16*)(ws + o); o += (size_t)Nn * 128 * 2;  //  4 MB
    _Float16* aggh = (_Float16*)(ws + o); o += (size_t)Nn * 128 * 2;  //  4 MB
    float*    agg  = (float*)   (ws + o); o += (size_t)Nn * 128 * 4;  //  8 MB
    // packed weights: size = NF*KT*512 halves  (NF = ceil(N/16), KT = K/32)
    const size_t szEnc = (size_t)32 * 16 * 512;  // N=512,K=512
    const size_t szLat = (size_t)8  * 16 * 512;  // N=120->128, K=512
    const size_t szHop = (size_t)8  * 4  * 512;  // N=128/120, K=128
    _Float16* pw1    = (_Float16*)(ws + o); o += szEnc * 2;
    _Float16* pw2    = (_Float16*)(ws + o); o += szEnc * 2;
    _Float16* pw3    = (_Float16*)(ws + o); o += szEnc * 2;
    _Float16* pw4    = (_Float16*)(ws + o); o += szLat * 2;
    _Float16* prelh  = (_Float16*)(ws + o); o += 4 * szHop * 2;
    _Float16* prooth = (_Float16*)(ws + o); o += 4 * szHop * 2;
    _Float16* prelf  = (_Float16*)(ws + o); o += szHop * 2;
    _Float16* prootf = (_Float16*)(ws + o); o += szHop * 2;

    dim3 blk(256);

    // ---- pack all weights (tiny; stays hot in 192MB L2)
    pack_w_kernel<<<(32 * 16 + 7) / 8, blk, 0, stream>>>(w1, pw1, 512, 512, 32 * 16);
    pack_w_kernel<<<(32 * 16 + 7) / 8, blk, 0, stream>>>(w2, pw2, 512, 512, 32 * 16);
    pack_w_kernel<<<(32 * 16 + 7) / 8, blk, 0, stream>>>(w3, pw3, 512, 512, 32 * 16);
    pack_w_kernel<<<(8 * 16 + 7) / 8,  blk, 0, stream>>>(w4, pw4, 512, 120, 8 * 16);
    for (int hop = 0; hop < 4; ++hop) {
        pack_w_kernel<<<(8 * 4 + 7) / 8, blk, 0, stream>>>(
            rel_w_h + (size_t)hop * 128 * 128, prelh + (size_t)hop * szHop, 128, 128, 8 * 4);
        pack_w_kernel<<<(8 * 4 + 7) / 8, blk, 0, stream>>>(
            root_w_h + (size_t)hop * 128 * 128, prooth + (size_t)hop * szHop, 128, 128, 8 * 4);
    }
    pack_w_kernel<<<(8 * 4 + 7) / 8, blk, 0, stream>>>(rel_w_f,  prelf,  128, 120, 8 * 4);
    pack_w_kernel<<<(8 * 4 + 7) / 8, blk, 0, stream>>>(root_w_f, prootf, 128, 120, 8 * 4);

    // ---- x -> f16
    f32_to_f16_kernel<<<(Nn * 512 / 4 + 255) / 256, blk, 0, stream>>>(x, xh, (long)Nn * 512);

    // ---- encoder MLP (f16 activations)
    dim3 gEnc(8, Nn / 128);   // N=512 -> 8 n-blocks of 64
    wmma_gemm_async<<<gEnc, blk, 0, stream>>>(xh,   pw1, nullptr, nullptr, b1, nullptr, act0, Nn, 512, 512, 1);
    wmma_gemm_async<<<gEnc, blk, 0, stream>>>(act0, pw2, nullptr, nullptr, b2, nullptr, act1, Nn, 512, 512, 1);
    wmma_gemm_async<<<gEnc, blk, 0, stream>>>(act1, pw3, nullptr, nullptr, b3, nullptr, act0, Nn, 512, 512, 1);
    dim3 gLat(2, Nn / 128);   // Npad=128 -> 2 n-blocks
    wmma_gemm_async<<<gLat, blk, 0, stream>>>(act0, pw4, nullptr, nullptr, b4, nullptr, act1, Nn, 120, 512, 0);

    // ---- concat [h | pos | u[batch]] -> 128 f16 features
    assemble_feat_kernel<<<(Nn * 128 + 255) / 256, blk, 0, stream>>>(act1, pos, u, batch, feat, Nn);

    const int* srcI = eidx;
    const int* dstI = eidx + E_;
    dim3 gScat((E_ + 7) / 8);
    dim3 gHop(2, Nn / 128);

    _Float16* hcur = feat;
    _Float16* hoth = hnxt;
    for (int hop = 0; hop < 4; ++hop) {
        hipMemsetAsync(agg, 0, (size_t)Nn * 128 * 4, stream);
        edge_scatter_kernel<<<gScat, blk, 0, stream>>>(hcur, ew, srcI, dstI, agg, E_);
        f32_to_f16_kernel<<<(Nn * 128 / 4 + 255) / 256, blk, 0, stream>>>(agg, aggh, (long)Nn * 128);
        // relu( agg@w_rel + h@w_root + b ) — fused dual-pass WMMA GEMM
        wmma_gemm_async<<<gHop, blk, 0, stream>>>(
            aggh, prelh + (size_t)hop * szHop,
            hcur, prooth + (size_t)hop * szHop,
            rel_b_h + (size_t)hop * 128,
            nullptr, hoth, Nn, 128, 128, 1);
        _Float16* t = hcur; hcur = hoth; hoth = t;
    }

    // ---- final gconv (no relu), f32 output 16384x120
    hipMemsetAsync(agg, 0, (size_t)Nn * 128 * 4, stream);
    edge_scatter_kernel<<<gScat, blk, 0, stream>>>(hcur, ew, srcI, dstI, agg, E_);
    f32_to_f16_kernel<<<(Nn * 128 / 4 + 255) / 256, blk, 0, stream>>>(agg, aggh, (long)Nn * 128);
    wmma_gemm_async<<<gLat, blk, 0, stream>>>(
        aggh, prelf, hcur, prootf, rel_b_f, (float*)d_out, nullptr, Nn, 120, 128, 0);
}
